// ReTiNDoA_77730318123085
// MI455X (gfx1250) — compile-verified, matrix operand order checked by reading, not run on hardware
//
#include <hip/hip_runtime.h>

// ---------------------------------------------------------------------------
// MI455X (gfx1250, wave32) pipeline for the pooled-MLP + gated-scan reference.
//   pool:  x(64,12,512,256) -> xf16 (32768 x 32)   [mean of ch0..7, 8-freq boxes]
//   G1:    xf @ W1 + b1, silu   -> h1 (32768x512) f16   (direct WMMA)
//   G2:    h1 @ W2 + b2, silu   -> h2 (32768x512) f16   (async-LDS staged WMMA)
//   G3:    h2 @ W3 + b3         -> z  (32768x96)  f16
//   G4:    silu(z @ Wf)         -> xproj (32768x96) f32
//   gate:  srp stats -> alpha (32768x96) f32
//   scan:  h_t = a*h + (1-a)*xp -> hs f16, hT f32
//   G5:    hs @ [Wh|Wo|0] + b   -> logits / delta directly into d_out
// ---------------------------------------------------------------------------

typedef __attribute__((ext_vector_type(16))) _Float16 v16h;
typedef __attribute__((ext_vector_type(8)))  _Float16 v8h;
typedef __attribute__((ext_vector_type(8)))  float    v8f;

#define MROWS 32768   // B*T = 64*512

__device__ __forceinline__ float sigmoidf_(float x) { return 1.0f / (1.0f + __expf(-x)); }
__device__ __forceinline__ float siluf_(float x)    { return x * sigmoidf_(x); }

// One WMMA fragment (A-style or Bt-style) from row-major f16: lane l<16 ->
// row (l&15), K halves [kb..kb+7] and [kb+16..kb+23]; lane>=16 -> kb=8.
__device__ __forceinline__ v16h load_frag16(const _Float16* __restrict__ base,
                                            int row, int ldk, int kk, int kb) {
  const _Float16* p = base + (size_t)row * ldk + kk + kb;
  union { v16h v; v8h h[2]; } u;
  u.h[0] = *(const v8h*)(p);
  u.h[1] = *(const v8h*)(p + 16);
  return u.v;
}

// ---------------- pooling: mean over x ch 0..7, 8-wide freq boxes ----------
__global__ void pool_kernel(const float* __restrict__ x, _Float16* __restrict__ xf) {
  int wave = threadIdx.x >> 5;
  int lane = threadIdx.x & 31;            // lane == output freq bin (0..31)
  int gw   = blockIdx.x * 8 + wave;       // (b,t) pair, 0..32767
  int b = gw >> 9, t = gw & 511;
  const float* base = x + (((size_t)b * 12) * 512 + t) * 256;
  float s = 0.f;
#pragma unroll
  for (int c = 0; c < 8; ++c) {
    const float4* p = (const float4*)(base + (size_t)c * (512 * 256) + lane * 8);
    float4 u = p[0], v = p[1];
    s += u.x + u.y + u.z + u.w + v.x + v.y + v.z + v.w;
  }
  xf[(size_t)gw * 32 + lane] = (_Float16)(s * (1.0f / 64.0f));
}

// ------------- weight prep: f32 [K][N] -> f16 transposed [Npad][K] ---------
__global__ void wprep_kernel(const float* __restrict__ src, _Float16* __restrict__ dst,
                             int K, int N, int Npad) {
  int id = blockIdx.x * 256 + threadIdx.x;
  if (id >= Npad * K) return;
  int n = id / K, k = id - n * K;
  float v = (n < N) ? src[(size_t)k * N + n] : 0.f;
  dst[id] = (_Float16)v;
}

// ------------- head prep: [Wh | Wo | zero-pad] transposed, fused bias ------
__global__ void headprep_kernel(const float* __restrict__ Wh, const float* __restrict__ Wo,
                                const float* __restrict__ bh, const float* __restrict__ bo,
                                _Float16* __restrict__ Wct, float* __restrict__ biasC) {
  int id = blockIdx.x * 256 + threadIdx.x;
  if (id < 96 * 96) {
    int n = id / 96, k = id - n * 96;     // dst[n][k]
    float v = (n < 72) ? Wh[(size_t)k * 72 + n] : ((n == 72) ? Wo[k] : 0.f);
    Wct[id] = (_Float16)v;
  }
  if (id < 96) biasC[id] = (id < 72) ? bh[id] : ((id == 72) ? bo[0] : 0.f);
}

// ---------------- generic direct-load WMMA GEMM ----------------------------
// Block: 256 threads = 8 waves; block tile 128(M) x 32(N); wave = one m-tile,
// two n-tiles (A fragment reused in registers). K multiple of 32.
enum { EPI_F16_SILU = 0, EPI_F16_LIN = 1, EPI_F32_SILU = 2, EPI_LOGITS = 3 };

template <int MODE>
__global__ void gemm_wmma_kernel(const _Float16* __restrict__ A,
                                 const _Float16* __restrict__ Bt,
                                 const float* __restrict__ bias,
                                 void* __restrict__ outp,
                                 float* __restrict__ delta,
                                 int N, int K) {
  int wave  = threadIdx.x >> 5;
  int lane  = threadIdx.x & 31;
  int mtile = blockIdx.x * 128 + wave * 16;
  int nb    = blockIdx.y * 32;
  int arow  = mtile + (lane & 15);
  int kb    = (lane >> 4) * 8;
  int brow0 = nb + (lane & 15);
  int brow1 = brow0 + 16;

  v8f acc0 = {}; v8f acc1 = {};
  for (int kk = 0; kk < K; kk += 32) {
    v16h a  = load_frag16(A,  arow,  K, kk, kb);
    v16h b0 = load_frag16(Bt, brow0, K, kk, kb);
    v16h b1 = load_frag16(Bt, brow1, K, kk, kb);
    acc0 = __builtin_amdgcn_wmma_f32_16x16x32_f16(false, a, false, b0, (short)0, acc0, false, false);
    acc1 = __builtin_amdgcn_wmma_f32_16x16x32_f16(false, a, false, b1, (short)0, acc1, false, false);
  }

  int mrow = mtile + ((lane >> 4) << 3);  // C/D layout: VGPR r -> M = r + 8*(lane/16)
  int ncol = lane & 15;                   //             N = lane % 16
#pragma unroll
  for (int r = 0; r < 8; ++r) {
    int m = mrow + r;
#pragma unroll
    for (int tile = 0; tile < 2; ++tile) {
      int n = nb + tile * 16 + ncol;
      float v = (tile == 0) ? acc0[r] : acc1[r];
      if (bias) v += bias[n];
      if constexpr (MODE == EPI_F16_SILU) {
        ((_Float16*)outp)[(size_t)m * N + n] = (_Float16)siluf_(v);
      } else if constexpr (MODE == EPI_F16_LIN) {
        ((_Float16*)outp)[(size_t)m * N + n] = (_Float16)v;
      } else if constexpr (MODE == EPI_F32_SILU) {
        ((float*)outp)[(size_t)m * N + n] = siluf_(v);
      } else {  // EPI_LOGITS: n<72 -> logits[m*72+n]; n==72 -> delta[m]; else pad
        if (n < 72)       ((float*)outp)[(size_t)m * 72 + n] = v;
        else if (n == 72) delta[m] = v;
      }
    }
  }
}

// ---------------- G2: async-LDS staged, double-buffered WMMA GEMM ----------
// Fixed shape: M=32768, N=512, K=512 (h1 @ W2t + b2, silu -> h2, f16 out).
// Block tile 128(M) x 64(N); 8 waves, each wave = 1 m-tile x 4 n-tiles.
// Per k-step (32): A tile 128x32 f16 (8KB) + B tile 64x32 f16 (4KB) staged to
// LDS with global_load_async_to_lds_b128 (ASYNCcnt), double buffered.
// LDS rows padded to 80B -> conflict-free ds_load_b128 fragment reads.
#define G2K     512
#define G2N     512
#define ASTRIDE 80                          // padded LDS row stride (bytes)
#define BUFSZ   ((128 + 64) * ASTRIDE)      // 15360 bytes per buffer

__global__ void __launch_bounds__(256)
gemm512_async_kernel(const _Float16* __restrict__ A, const _Float16* __restrict__ Bt,
                     const float* __restrict__ bias, _Float16* __restrict__ out) {
  __shared__ __align__(16) char smem[2 * BUFSZ];
  const unsigned ldsBase = (unsigned)(uintptr_t)(&smem[0]);  // low 32b = LDS offset

  const int t    = threadIdx.x;
  const int wave = t >> 5;
  const int lane = t & 31;
  const int mbase = blockIdx.x * 128;
  const int nbase = blockIdx.y * 64;

  // Staging roles per thread: A rows (t>>2) and 64+(t>>2), B row (t>>2);
  // 16-byte segment (t&3) of each 64-byte row chunk.  3 async b128 per step.
  const int srow = t >> 2;
  const int sseg = (t & 3) * 16;
  const uint64_t aBase = (uint64_t)(uintptr_t)A;
  const uint64_t bBase = (uint64_t)(uintptr_t)Bt;

  auto issue = [&](int buf, int kk) {
    unsigned lds0 = ldsBase + buf * BUFSZ;
    uint64_t ga0 = aBase + ((size_t)(mbase + srow) * G2K + kk) * 2 + sseg;
    uint64_t ga1 = aBase + ((size_t)(mbase + 64 + srow) * G2K + kk) * 2 + sseg;
    uint64_t gb  = bBase + ((size_t)(nbase + srow) * G2K + kk) * 2 + sseg;
    unsigned la0 = lds0 + srow * ASTRIDE + sseg;
    unsigned la1 = lds0 + (64 + srow) * ASTRIDE + sseg;
    unsigned lb  = lds0 + 128 * ASTRIDE + srow * ASTRIDE + sseg;
    asm volatile("global_load_async_to_lds_b128 %0, %1, off" :: "v"(la0), "v"(ga0) : "memory");
    asm volatile("global_load_async_to_lds_b128 %0, %1, off" :: "v"(la1), "v"(ga1) : "memory");
    asm volatile("global_load_async_to_lds_b128 %0, %1, off" :: "v"(lb),  "v"(gb)  : "memory");
  };

  const int frow = lane & 15;
  const int kbB  = (lane >> 4) * 16;  // byte offset of first 8-half K chunk
  v8f acc[4] = {};

  issue(0, 0);
  for (int s = 0; s < G2K / 32; ++s) {
    const int cur = s & 1;
    if (s + 1 < G2K / 32) {
      issue(cur ^ 1, (s + 1) * 32);
      // 6 outstanding per wave; async loads complete in order, so <=3 means
      // the current buffer's 3 copies have landed in LDS.
      asm volatile("s_wait_asynccnt 0x3" ::: "memory");
    } else {
      asm volatile("s_wait_asynccnt 0x0" ::: "memory");
    }
    __syncthreads();  // all waves' staged data visible

    const char* base = smem + cur * BUFSZ;
    unsigned aoff = (unsigned)((wave * 16 + frow) * ASTRIDE + kbB);
    union { v16h v; v8h h[2]; } ua;
    ua.h[0] = *(const v8h*)(base + aoff);
    ua.h[1] = *(const v8h*)(base + aoff + 32);
#pragma unroll
    for (int nt = 0; nt < 4; ++nt) {
      unsigned boff = (unsigned)(128 * ASTRIDE + (nt * 16 + frow) * ASTRIDE + kbB);
      union { v16h v; v8h h[2]; } ub;
      ub.h[0] = *(const v8h*)(base + boff);
      ub.h[1] = *(const v8h*)(base + boff + 32);
      acc[nt] = __builtin_amdgcn_wmma_f32_16x16x32_f16(false, ua.v, false, ub.v, (short)0,
                                                       acc[nt], false, false);
    }
    __syncthreads();  // done reading cur before it is overwritten next iter
  }

  const int mrow = mbase + wave * 16 + ((lane >> 4) << 3);
  const int ncol = lane & 15;
#pragma unroll
  for (int nt = 0; nt < 4; ++nt) {
#pragma unroll
    for (int r = 0; r < 8; ++r) {
      int m = mrow + r;
      int n = nbase + nt * 16 + ncol;
      float v = acc[nt][r] + bias[n];
      out[(size_t)m * G2N + n] = (_Float16)siluf_(v);
    }
  }
}

// ---------------- gate stats: q = srp/sum, peak, curvature -> sigmoid ------
__global__ void alpha_kernel(const float* __restrict__ srp, const float* __restrict__ Wg,
                             const float* __restrict__ bg, float* __restrict__ alpha) {
  __shared__ float q[72];
  __shared__ float stats[2];
  int bt = blockIdx.x;
  int i  = threadIdx.x;     // blockDim = 96
  const float* s = srp + (size_t)bt * 72;
  if (i < 72) q[i] = s[i];
  __syncthreads();
  if (i == 0) {
    float sum = 0.f, mx = -1e30f;
    for (int j = 0; j < 72; ++j) { sum += q[j]; mx = fmaxf(mx, q[j]); }
    float inv = 1.0f / (sum + 1e-8f);
    float c = 0.f;
    for (int j = 0; j < 72; ++j) {
      float qm = q[(j + 71) % 72], q0 = q[j], qp = q[(j + 1) % 72];
      c += fabsf((qm - 2.f * q0 + qp) * inv);
    }
    stats[0] = mx * inv;
    stats[1] = c * (1.0f / 72.0f);
  }
  __syncthreads();
  if (i < 96) {
    float v = stats[0] * Wg[i] + stats[1] * Wg[96 + i] + bg[i];
    alpha[(size_t)bt * 96 + i] = sigmoidf_(v);
  }
}

// ---------------- sequential gated scan over T ----------------------------
__global__ void scan_kernel(const float* __restrict__ alpha, const float* __restrict__ xproj,
                            _Float16* __restrict__ hs, float* __restrict__ hT) {
  int b = blockIdx.x;       // 64 blocks
  int c = threadIdx.x;      // 96 channels
  float h = 0.f;
  size_t base = (size_t)b * 512 * 96 + c;
  for (int t = 0; t < 512; ++t) {
    size_t idx = base + (size_t)t * 96;
    float a  = alpha[idx];
    float xp = xproj[idx];
    h = a * h + (1.f - a) * xp;
    hs[idx] = (_Float16)h;
  }
  hT[b * 96 + c] = h;
}

// ---------------------------------------------------------------------------
extern "C" void kernel_launch(void* const* d_in, const int* in_sizes, int n_in,
                              void* d_out, int out_size, void* d_ws, size_t ws_size,
                              hipStream_t stream) {
  const float* x   = (const float*)d_in[0];
  const float* srp = (const float*)d_in[1];
  const float* W1  = (const float*)d_in[2];
  const float* b1  = (const float*)d_in[3];
  const float* W2  = (const float*)d_in[4];
  const float* b2  = (const float*)d_in[5];
  const float* W3  = (const float*)d_in[6];
  const float* b3  = (const float*)d_in[7];
  const float* Wf  = (const float*)d_in[8];
  const float* Wg  = (const float*)d_in[9];
  const float* bg  = (const float*)d_in[10];
  const float* Wh  = (const float*)d_in[11];
  const float* bh  = (const float*)d_in[12];
  const float* Wo  = (const float*)d_in[13];
  const float* bo  = (const float*)d_in[14];

  char* ws = (char*)d_ws;
  size_t off = 0;
  auto take = [&](size_t bytes) -> char* {
    char* p = ws + off;
    off = (off + bytes + 255) & ~(size_t)255;
    return p;
  };
  _Float16* W1t = (_Float16*)take(512 * 32 * 2);
  _Float16* W2t = (_Float16*)take(512 * 512 * 2);
  _Float16* W3t = (_Float16*)take(96 * 512 * 2);
  _Float16* Wft = (_Float16*)take(96 * 96 * 2);
  _Float16* Wct = (_Float16*)take(96 * 96 * 2);
  float*    bC  = (float*)take(96 * 4);
  _Float16* xf  = (_Float16*)take((size_t)MROWS * 32 * 2);
  _Float16* h1  = (_Float16*)take((size_t)MROWS * 512 * 2);
  _Float16* h2  = (_Float16*)take((size_t)MROWS * 512 * 2);
  _Float16* z   = (_Float16*)take((size_t)MROWS * 96 * 2);
  float*    alp = (float*)take((size_t)MROWS * 96 * 4);
  float*    xpj = (float*)take((size_t)MROWS * 96 * 4);
  _Float16* hs  = (_Float16*)take((size_t)MROWS * 96 * 2);

  float* outF   = (float*)d_out;
  float* logits = outF;                                  // (32768, 72)
  float* delta  = outF + (size_t)MROWS * 72;             // (32768, 1)
  float* hT     = delta + MROWS;                         // (64, 96)

  // Weight conversion/transposition (runs inside the captured graph; cheap).
  wprep_kernel<<<(512 * 32 + 255) / 256, 256, 0, stream>>>(W1, W1t, 32, 512, 512);
  wprep_kernel<<<(512 * 512 + 255) / 256, 256, 0, stream>>>(W2, W2t, 512, 512, 512);
  wprep_kernel<<<(96 * 512 + 255) / 256, 256, 0, stream>>>(W3, W3t, 512, 96, 96);
  wprep_kernel<<<(96 * 96 + 255) / 256, 256, 0, stream>>>(Wf, Wft, 96, 96, 96);
  headprep_kernel<<<(96 * 96 + 255) / 256, 256, 0, stream>>>(Wh, Wo, bh, bo, Wct, bC);

  pool_kernel<<<MROWS / 8, 256, 0, stream>>>(x, xf);

  gemm_wmma_kernel<EPI_F16_SILU><<<dim3(MROWS / 128, 512 / 32), 256, 0, stream>>>(
      xf, W1t, b1, h1, nullptr, 512, 32);
  gemm512_async_kernel<<<dim3(MROWS / 128, 512 / 64), 256, 0, stream>>>(h1, W2t, b2, h2);
  gemm_wmma_kernel<EPI_F16_LIN><<<dim3(MROWS / 128, 96 / 32), 256, 0, stream>>>(
      h2, W3t, b3, z, nullptr, 96, 512);
  gemm_wmma_kernel<EPI_F32_SILU><<<dim3(MROWS / 128, 96 / 32), 256, 0, stream>>>(
      z, Wft, nullptr, xpj, nullptr, 96, 96);

  alpha_kernel<<<MROWS, 96, 0, stream>>>(srp, Wg, bg, alp);
  scan_kernel<<<64, 96, 0, stream>>>(alp, xpj, hs, hT);

  gemm_wmma_kernel<EPI_LOGITS><<<dim3(MROWS / 128, 96 / 32), 256, 0, stream>>>(
      hs, Wct, bC, logits, delta, 96, 96);
}
